// VisionMamba_79860621902568
// MI455X (gfx1250) — matrix operand order repaired and checked by reading, not compiled, not run on hardware
//
#include <hip/hip_runtime.h>
#include <hip/hip_bf16.h>
#include <math.h>
#include <stdint.h>

// ---------------- model constants ----------------
constexpr int DEPTH   = 24;
constexpr int DM      = 192;   // d_model
constexpr int DI      = 384;   // d_inner
constexpr int DS      = 16;    // d_state
constexpr int RK      = 12;    // dt_rank
constexpr int XD      = RK + 2 * DS;   // 44
constexpr int XDP     = 48;            // padded ld for x_dbl
constexpr int Bb      = 2;
constexpr int Tt      = 8;
constexpr int NP      = 196;   // patches per frame
constexpr int L       = 1 + Tt * NP;   // 1569
constexpr int NTOK    = Bb * L;        // 3138

typedef __attribute__((ext_vector_type(16))) _Float16 v16h;
typedef __attribute__((ext_vector_type(8)))  _Float16 v8h;
typedef __attribute__((ext_vector_type(8)))  float    v8f;

#if defined(__HIP_DEVICE_COMPILE__) && defined(__gfx1250__) && \
    __has_builtin(__builtin_amdgcn_global_load_async_to_lds_b32) && \
    __has_builtin(__builtin_amdgcn_s_wait_asynccnt)
#define USE_ASYNC_LDS 1
#else
#define USE_ASYNC_LDS 0
#endif

__device__ __forceinline__ float silu_f(float x)     { return x / (1.0f + expf(-x)); }
__device__ __forceinline__ float softplus_f(float x) { return (x > 20.0f) ? x : log1pf(expf(x)); }

// ---------------- f32 -> f16 bulk convert (weights, once per call) ----------------
__global__ void cvt16_kernel(const float* __restrict__ src, _Float16* __restrict__ dst, int n)
{
    int i = (int)(blockIdx.x * blockDim.x + threadIdx.x);
    if (i < n) dst[i] = (_Float16)src[i];
}

// ---------------- patch embed + positional: builds hid16 (f16) and res (f32) ----------------
__global__ void patch_embed_kernel(const float* __restrict__ x,
                                   const float* __restrict__ patch_w,
                                   const float* __restrict__ patch_b,
                                   const float* __restrict__ cls_token,
                                   const float* __restrict__ pos_embed,
                                   const float* __restrict__ temporal_pos,
                                   _Float16* __restrict__ hid16,
                                   float* __restrict__ res)
{
    int tok = blockIdx.x;          // b*L + s
    int c   = threadIdx.x;         // 0..191
    int b   = tok / L;
    int s   = tok - b * L;
    float v;
    if (s == 0) {
        v = cls_token[c] + pos_embed[c];
    } else {
        int q  = s - 1;
        int t  = q / NP;
        int p  = q - t * NP;
        int py = p / 14, px = p - py * 14;
        float acc = patch_b[c];
        for (int ch = 0; ch < 3; ++ch) {
            for (int i = 0; i < 16; ++i) {
                const float* xr = x + ((((size_t)b * 3 + ch) * Tt + t) * 224 + (py * 16 + i)) * 224 + px * 16;
                const float* wr = patch_w + (((size_t)c * 3 + ch) * 16 + i) * 16;
                #pragma unroll
                for (int j = 0; j < 16; ++j) acc += xr[j] * wr[j];
            }
        }
        v = acc + pos_embed[(size_t)(1 + p) * DM + c] + temporal_pos[(size_t)t * DM + c];
    }
    hid16[(size_t)tok * DM + c] = (_Float16)v;
    res[(size_t)tok * DM + c]   = v;
}

// ---------------- WMMA GEMM: C[M,N] = A[M,K] * W[N,K]^T,  A/W in f16, C in f32 ----------------
// MODE 0: C = acc
// MODE 2: C = softplus(acc + bias[n])
// MODE 3: C += acc
// MODE 4: C = acc; and for col < RK also C2[m*16+col] = (f16)acc   (dt columns)
template<int K, int MODE>
__global__ void gemm_wmma_kernel(const _Float16* __restrict__ A, int lda,
                                 const _Float16* __restrict__ W, int ldw,
                                 float* C, int ldc, int M, int N,
                                 const float* __restrict__ bias,
                                 _Float16* __restrict__ C2)
{
    int wave   = (int)((blockIdx.x * blockDim.x + threadIdx.x) >> 5);
    int lane   = (int)(threadIdx.x & 31);
    int mTiles = (M + 15) >> 4;
    int nTiles = (N + 15) >> 4;
    if (wave >= mTiles * nTiles) return;
    int tm = wave % mTiles;
    int tn = wave / mTiles;

    int row   = tm * 16 + (lane & 15);
    int wrow  = tn * 16 + (lane & 15);
    int khalf = lane >> 4;
    // Clamp: out-of-range rows/cols load valid finite data whose products land only
    // in tile rows/cols that are never stored.
    int rowc  = (row  < M) ? row  : (M - 1);
    int wrowc = (wrow < N) ? wrow : (N - 1);

    const _Float16* Abase = A + (size_t)rowc  * lda + khalf * 8;
    const _Float16* Wbase = W + (size_t)wrowc * ldw + khalf * 8;

    v8f acc = {};
    #pragma unroll
    for (int kk = 0; kk < K; kk += 32) {
        v16h a, bfr;
        if constexpr (K % 32 == 0) {
            v8h a0 = *(const v8h*)(Abase + kk);
            v8h a1 = *(const v8h*)(Abase + kk + 16);
            v8h b0 = *(const v8h*)(Wbase + kk);
            v8h b1 = *(const v8h*)(Wbase + kk + 16);
            #pragma unroll
            for (int e = 0; e < 8; ++e) {
                a[e] = a0[e];  a[e + 8]  = a1[e];
                bfr[e] = b0[e]; bfr[e + 8] = b1[e];
            }
        } else {
            // K=12 (dtproj): compile-time constant guards fold to load-12 / zero-4
            #pragma unroll
            for (int e = 0; e < 16; ++e) {
                int k = kk + ((e < 8) ? (khalf * 8 + e) : (16 + khalf * 8 + (e - 8)));
                a[e] = (k < K) ? A[(size_t)rowc * lda + k] : (_Float16)0.0f;
            }
            #pragma unroll
            for (int e = 0; e < 16; ++e) {
                int k = kk + ((e < 8) ? (khalf * 8 + e) : (16 + khalf * 8 + (e - 8)));
                bfr[e] = (k < K) ? W[(size_t)wrowc * ldw + k] : (_Float16)0.0f;
            }
        }
        acc = __builtin_amdgcn_wmma_f32_16x16x32_f16(false, a, false, bfr,
                                                     (short)0, acc, false, false);
    }

    int col = tn * 16 + (lane & 15);
    bool interior = ((tm * 16 + 15) < M) && ((tn * 16 + 15) < N);
    #pragma unroll
    for (int r = 0; r < 8; ++r) {
        int m = tm * 16 + khalf * 8 + r;
        if (interior || (m < M && col < N)) {
            float v = acc[r];
            if constexpr (MODE == 2) v = softplus_f(v + bias[col]);
            float* p = &C[(size_t)m * ldc + col];
            if constexpr (MODE == 3) *p += v; else *p = v;
            if constexpr (MODE == 4) {
                if (col < RK) C2[(size_t)m * 16 + col] = (_Float16)v;
            }
        }
    }
}

// ---------------- conv(fwd causal / bwd anti-causal) + silu; writes f32 and f16 copies ----------------
__global__ void conv_silu_kernel(const float* __restrict__ XZ,
                                 const float* __restrict__ cwf, const float* __restrict__ cbf,
                                 const float* __restrict__ cwb, const float* __restrict__ cbb,
                                 float* __restrict__ xcf,  float* __restrict__ xcb,
                                 _Float16* __restrict__ xcf16, _Float16* __restrict__ xcb16)
{
    int idx = (int)(blockIdx.x * blockDim.x + threadIdx.x);
    if (idx >= NTOK * DI) return;
    int d   = idx % DI;
    int tok = idx / DI;
    int b   = tok / L;
    int l   = tok - b * L;
    float accf = cbf[d], accb = cbb[d];
    #pragma unroll
    for (int k = 0; k < 4; ++k) {
        int lf = l - 3 + k;
        if (lf >= 0) accf += cwf[d * 4 + k] * XZ[((size_t)(b * L + lf)) * (2 * DI) + d];
        int lb = l + 3 - k;
        if (lb < L) accb += cwb[d * 4 + k] * XZ[((size_t)(b * L + lb)) * (2 * DI) + d];
    }
    float f = silu_f(accf), g = silu_f(accb);
    xcf[idx] = f;  xcb[idx] = g;
    xcf16[idx] = (_Float16)f;  xcb16[idx] = (_Float16)g;
}

// ---------------- selective scan; B/C staged into LDS via async-to-LDS (1 step ahead) ----------------
__global__ void scan_kernel(const float* xc, const float* __restrict__ dt,
                            const float* __restrict__ xdbl,
                            const float* __restrict__ A_log, const float* __restrict__ Dp,
                            float* y, int reverse)
{
    int d   = (int)(blockIdx.x * 128 + threadIdx.x);   // 3 blocks x 128 = 384 channels
    int b   = (int)blockIdx.y;
    int tid = (int)threadIdx.x;
    float a[DS], h[DS];
    #pragma unroll
    for (int n = 0; n < DS; ++n) { a[n] = -expf(A_log[(size_t)d * DS + n]); h[n] = 0.0f; }
    float Dpd = Dp[d];
    __shared__ float sBC[3][32];   // [slot][ B(0..15) | C(16..31) ]

#if USE_ASYNC_LDS
    typedef __attribute__((address_space(1))) int* as1i;   // global int*
    typedef __attribute__((address_space(3))) int* as3i;   // LDS int*
#endif

    // token index of a scan step
    #define TOK_OF(step) ((size_t)b * L + (reverse ? (L - 1 - (step)) : (step)))

    if (tid < 32) {
#if USE_ASYNC_LDS
        const float* g0 = xdbl + TOK_OF(0) * XDP + RK + tid;
        __builtin_amdgcn_global_load_async_to_lds_b32(
            (as1i)(unsigned long long)(uintptr_t)g0,
            (as3i)(unsigned int)(uintptr_t)&sBC[0][tid], 0, 0);
#else
        sBC[0][tid] = xdbl[TOK_OF(0) * XDP + RK + tid];
#endif
    }

    for (int step = 0; step < L; ++step) {
        int par = step % 3;
        if (tid < 32) {
            if (step + 1 < L) {
                int pn = (step + 1) % 3;
#if USE_ASYNC_LDS
                const float* g = xdbl + TOK_OF(step + 1) * XDP + RK + tid;
                __builtin_amdgcn_global_load_async_to_lds_b32(
                    (as1i)(unsigned long long)(uintptr_t)g,
                    (as3i)(unsigned int)(uintptr_t)&sBC[pn][tid], 0, 0);
                __builtin_amdgcn_s_wait_asynccnt(1);   // current step's load complete
#else
                sBC[pn][tid] = xdbl[TOK_OF(step + 1) * XDP + RK + tid];
#endif
            } else {
#if USE_ASYNC_LDS
                __builtin_amdgcn_s_wait_asynccnt(0);
#endif
            }
        }
        __syncthreads();
        size_t tok = TOK_OF(step);
        float dtv = dt[tok * DI + d];
        float u   = xc[tok * DI + d];
        float du  = dtv * u;
        float acc = 0.0f;
        #pragma unroll
        for (int n = 0; n < DS; ++n) {
            float dA = expf(dtv * a[n]);
            h[n] = dA * h[n] + du * sBC[par][n];
            acc += h[n] * sBC[par][DS + n];
        }
        y[tok * DI + d] = acc + u * Dpd;   // same thread read u above; in-place safe
    }
    #undef TOK_OF
}

// ---------------- gate: Y16 = (f16)((y_f + y_b) * silu(z)) ----------------
__global__ void combine_kernel(const float* __restrict__ yf, const float* __restrict__ yb,
                               const float* __restrict__ XZ, _Float16* __restrict__ Y16)
{
    int idx = (int)(blockIdx.x * blockDim.x + threadIdx.x);
    if (idx >= NTOK * DI) return;
    int d   = idx % DI;
    int tok = idx / DI;
    float z = XZ[(size_t)tok * (2 * DI) + DI + d];
    Y16[idx] = (_Float16)((yf[idx] + yb[idx]) * silu_f(z));
}

// ---------------- RMSNorm of res -> hid16 (f16) ----------------
__global__ void rmsnorm_kernel(const float* __restrict__ in, const float* __restrict__ w,
                               _Float16* __restrict__ out16)
{
    int tok = blockIdx.x;
    int c   = (int)threadIdx.x;   // 192 threads
    __shared__ float red[DM];
    float v = in[(size_t)tok * DM + c];
    red[c] = v * v;
    __syncthreads();
    for (int s = 96; s >= 3; s >>= 1) {
        if (c < s) red[c] += red[c + s];
        __syncthreads();
    }
    if (c == 0) red[0] = red[0] + red[1] + red[2];
    __syncthreads();
    float scale = rsqrtf(red[0] / (float)DM + 1e-5f);
    out16[(size_t)tok * DM + c] = (_Float16)(v * scale * w[c]);
}

// ---------------- final: hid=rmsnorm(res,nw_last); out = rmsnorm(hid+res,nfw) @ head ----------------
__global__ void head_kernel(const float* __restrict__ res, const float* __restrict__ nw_last,
                            const float* __restrict__ nfw,
                            const float* __restrict__ hw, const float* __restrict__ hb,
                            float* __restrict__ out)
{
    int c = (int)threadIdx.x;   // 192 threads, one block
    __shared__ float red[DM];
    __shared__ float xn[DM];
    for (int b = 0; b < Bb; ++b) {
        float rv = res[(size_t)b * L * DM + c];
        // hid = rmsnorm(res, norm_w[last])
        red[c] = rv * rv;
        __syncthreads();
        for (int s = 96; s >= 3; s >>= 1) {
            if (c < s) red[c] += red[c + s];
            __syncthreads();
        }
        if (c == 0) red[0] = red[0] + red[1] + red[2];
        __syncthreads();
        float scale1 = rsqrtf(red[0] / (float)DM + 1e-5f);
        float v = rv * scale1 * nw_last[c] + rv;          // hid + res
        __syncthreads();
        // out = rmsnorm(hid + res, norm_f_w)
        red[c] = v * v;
        __syncthreads();
        for (int s = 96; s >= 3; s >>= 1) {
            if (c < s) red[c] += red[c + s];
            __syncthreads();
        }
        if (c == 0) red[0] = red[0] + red[1] + red[2];
        __syncthreads();
        float scale2 = rsqrtf(red[0] / (float)DM + 1e-5f);
        xn[c] = v * scale2 * nfw[c];
        __syncthreads();
        if (c < 2) {
            float acc = hb[c];
            for (int k = 0; k < DM; ++k) acc += xn[k] * hw[(size_t)c * DM + k];
            out[b * 2 + c] = acc;
        }
        __syncthreads();
    }
}

// ---------------- host launch ----------------
static inline int gemm_grid(int M, int N) {
    int tiles = ((M + 15) / 16) * ((N + 15) / 16);
    return (tiles + 7) / 8;   // 8 wave32 per 256-thread block
}

extern "C" void kernel_launch(void* const* d_in, const int* in_sizes, int n_in,
                              void* d_out, int out_size, void* d_ws, size_t ws_size,
                              hipStream_t stream)
{
    const float* x            = (const float*)d_in[0];
    const float* patch_w      = (const float*)d_in[1];
    const float* patch_b      = (const float*)d_in[2];
    const float* cls_token    = (const float*)d_in[3];
    const float* pos_embed    = (const float*)d_in[4];
    const float* temporal_pos = (const float*)d_in[5];
    const float* in_proj_w    = (const float*)d_in[6];
    const float* conv_w       = (const float*)d_in[7];
    const float* conv_b       = (const float*)d_in[8];
    const float* xproj_w      = (const float*)d_in[9];
    const float* dtproj_w     = (const float*)d_in[10];
    const float* dtproj_b     = (const float*)d_in[11];
    const float* A_log        = (const float*)d_in[12];
    const float* Dp           = (const float*)d_in[13];
    const float* conv_w_b     = (const float*)d_in[14];
    const float* conv_b_b     = (const float*)d_in[15];
    const float* xproj_w_b    = (const float*)d_in[16];
    const float* dtproj_w_b   = (const float*)d_in[17];
    const float* dtproj_b_b   = (const float*)d_in[18];
    const float* A_log_b      = (const float*)d_in[19];
    const float* Dp_b         = (const float*)d_in[20];
    const float* out_proj_w   = (const float*)d_in[21];
    const float* norm_w       = (const float*)d_in[22];
    const float* norm_f_w     = (const float*)d_in[23];
    const float* head_w       = (const float*)d_in[24];
    const float* head_b       = (const float*)d_in[25];
    (void)in_sizes; (void)n_in; (void)out_size; (void)ws_size;

    // ---- workspace carve (256B aligned slabs) ----
    char* cur = (char*)d_ws;
    auto af32 = [&](size_t n) { float* p = (float*)cur;       cur += ((n * 4 + 255) / 256) * 256; return p; };
    auto af16 = [&](size_t n) { _Float16* p = (_Float16*)cur; cur += ((n * 2 + 255) / 256) * 256; return p; };

    float*    XZ      = af32((size_t)NTOK * 2 * DI);
    float*    xcf     = af32((size_t)NTOK * DI);
    float*    xcb     = af32((size_t)NTOK * DI);
    float*    xdbl_f  = af32((size_t)NTOK * XDP);
    float*    xdbl_b  = af32((size_t)NTOK * XDP);
    float*    dt_f    = af32((size_t)NTOK * DI);
    float*    dt_b    = af32((size_t)NTOK * DI);
    float*    res     = af32((size_t)NTOK * DM);
    _Float16* hid16   = af16((size_t)NTOK * DM);
    _Float16* xcf16   = af16((size_t)NTOK * DI);
    _Float16* xcb16   = af16((size_t)NTOK * DI);
    _Float16* xdt16_f = af16((size_t)NTOK * 16);
    _Float16* xdt16_b = af16((size_t)NTOK * 16);
    _Float16* Y16     = af16((size_t)NTOK * DI);
    _Float16* w16_ip  = af16((size_t)DEPTH * 2 * DI * DM);
    _Float16* w16_xpf = af16((size_t)DEPTH * XD * DI);
    _Float16* w16_xpb = af16((size_t)DEPTH * XD * DI);
    _Float16* w16_dtf = af16((size_t)DEPTH * DI * RK);
    _Float16* w16_dtb = af16((size_t)DEPTH * DI * RK);
    _Float16* w16_op  = af16((size_t)DEPTH * DM * DI);

    // ---- one-time (per call) weight conversion to f16 ----
    auto cvt = [&](const float* s, _Float16* d, int n) {
        cvt16_kernel<<<(n + 255) / 256, 256, 0, stream>>>(s, d, n);
    };
    cvt(in_proj_w,  w16_ip,  DEPTH * 2 * DI * DM);
    cvt(xproj_w,    w16_xpf, DEPTH * XD * DI);
    cvt(xproj_w_b,  w16_xpb, DEPTH * XD * DI);
    cvt(dtproj_w,   w16_dtf, DEPTH * DI * RK);
    cvt(dtproj_w_b, w16_dtb, DEPTH * DI * RK);
    cvt(out_proj_w, w16_op,  DEPTH * DM * DI);

    int eltGrid = (NTOK * DI + 255) / 256;

    patch_embed_kernel<<<NTOK, DM, 0, stream>>>(x, patch_w, patch_b, cls_token,
                                                pos_embed, temporal_pos, hid16, res);

    for (int layer = 0; layer < DEPTH; ++layer) {
        const _Float16* ipw = w16_ip  + (size_t)layer * 2 * DI * DM;
        const _Float16* xpf = w16_xpf + (size_t)layer * XD * DI;
        const _Float16* xpb = w16_xpb + (size_t)layer * XD * DI;
        const _Float16* dwf = w16_dtf + (size_t)layer * DI * RK;
        const _Float16* dwb = w16_dtb + (size_t)layer * DI * RK;
        const _Float16* opw = w16_op  + (size_t)layer * DM * DI;
        const float* cwf = conv_w     + (size_t)layer * DI * 4;
        const float* cbf = conv_b     + (size_t)layer * DI;
        const float* dbf = dtproj_b   + (size_t)layer * DI;
        const float* alf = A_log      + (size_t)layer * DI * DS;
        const float* dpf = Dp         + (size_t)layer * DI;
        const float* cwb = conv_w_b   + (size_t)layer * DI * 4;
        const float* cbb = conv_b_b   + (size_t)layer * DI;
        const float* dbb = dtproj_b_b + (size_t)layer * DI;
        const float* alb = A_log_b    + (size_t)layer * DI * DS;
        const float* dpb = Dp_b       + (size_t)layer * DI;
        const float* nw  = norm_w     + (size_t)layer * DM;

        // xz = hid @ in_proj_w^T              (M=3138, N=768, K=192)
        gemm_wmma_kernel<DM, 0><<<gemm_grid(NTOK, 2 * DI), 256, 0, stream>>>(
            hid16, DM, ipw, DM, XZ, 2 * DI, NTOK, 2 * DI, nullptr, nullptr);

        // both-direction depthwise conv + silu (f32 for scan, f16 for xproj GEMM)
        conv_silu_kernel<<<eltGrid, 256, 0, stream>>>(XZ, cwf, cbf, cwb, cbb,
                                                      xcf, xcb, xcf16, xcb16);

        // x_dbl = xc @ xproj_w^T  (N=44, K=384); dual-store dt columns as f16
        gemm_wmma_kernel<DI, 4><<<gemm_grid(NTOK, XD), 256, 0, stream>>>(
            xcf16, DI, xpf, DI, xdbl_f, XDP, NTOK, XD, nullptr, xdt16_f);
        gemm_wmma_kernel<DI, 4><<<gemm_grid(NTOK, XD), 256, 0, stream>>>(
            xcb16, DI, xpb, DI, xdbl_b, XDP, NTOK, XD, nullptr, xdt16_b);

        // dt = softplus(dt_in @ dtproj_w^T + dtproj_b)   (K=12 zero-padded at compile time)
        gemm_wmma_kernel<RK, 2><<<gemm_grid(NTOK, DI), 256, 0, stream>>>(
            xdt16_f, 16, dwf, RK, dt_f, DI, NTOK, DI, dbf, nullptr);
        gemm_wmma_kernel<RK, 2><<<gemm_grid(NTOK, DI), 256, 0, stream>>>(
            xdt16_b, 16, dwb, RK, dt_b, DI, NTOK, DI, dbb, nullptr);

        // selective scans (y written in place over xc f32 buffers)
        scan_kernel<<<dim3(DI / 128, Bb), 128, 0, stream>>>(xcf, dt_f, xdbl_f, alf, dpf, xcf, 0);
        scan_kernel<<<dim3(DI / 128, Bb), 128, 0, stream>>>(xcb, dt_b, xdbl_b, alb, dpb, xcb, 1);

        // Y16 = (y_f + y_b) * silu(z)
        combine_kernel<<<eltGrid, 256, 0, stream>>>(xcf, xcb, XZ, Y16);

        // res += Y @ out_proj_w^T             (M=3138, N=192, K=384)
        gemm_wmma_kernel<DI, 3><<<gemm_grid(NTOK, DM), 256, 0, stream>>>(
            Y16, DI, opw, DI, res, DM, NTOK, DM, nullptr, nullptr);

        // hid16 = (f16) rmsnorm(res, norm_w[layer])
        rmsnorm_kernel<<<NTOK, DM, 0, stream>>>(res, nw, hid16);
    }

    head_kernel<<<1, DM, 0, stream>>>(res, norm_w + (size_t)(DEPTH - 1) * DM,
                                      norm_f_w, head_w, head_b, (float*)d_out);
}
